// DeepseekV3Attention_72000831750707
// MI455X (gfx1250) — matrix operand order repaired and checked by reading, not compiled, not run on hardware
//
#include <hip/hip_runtime.h>
#include <hip/hip_bf16.h>
#include <hip/hip_fp16.h>

// ---------------------------------------------------------------------------
// DeepSeek-V3 MLA prefill for MI455X (gfx1250, wave32, WMMA 16x16x32 f16,
// async global->LDS weight staging tracked by ASYNCcnt)
// ---------------------------------------------------------------------------

typedef __attribute__((ext_vector_type(16))) _Float16 v16h;
typedef __attribute__((ext_vector_type(8)))  float    v8f;

#define S_LEN 2048
#define D_MODEL 2048
#define Q_LORA 1536
#define KV_LORA 512
#define ROPE_D 64
#define NHEAD 16
#define NOPE_D 128
#define V_D 128
#define QHD 192   // NOPE + ROPE

// --------------------------- WMMA fragment helpers --------------------------
// A fragment: 16(M) x 32(K) f16 tile, rows k-major with leading dim ld halves.
// ISA 7.12.2: lanes 0-15 hold M=0..15 with K={0..7,16..23}; lanes 16-31 hold
// the same rows with K={8..15,24..31}.
__device__ __forceinline__ v16h load_frag_a(const _Float16* base, int ld) {
  int lane = threadIdx.x & 31;
  int grp  = lane >> 4;
  int r    = lane & 15;
  const _Float16* p = base + (size_t)r * ld + (grp ? 8 : 0);
  v16h f;
  ((uint4*)&f)[0] = *(const uint4*)(p);
  ((uint4*)&f)[1] = *(const uint4*)(p + 16);
  return f;
}

// B fragment: 32(K) x 16(N) f16 tile taken from W stored as N rows x K cols
// (k-major per output column, i.e. the w.T pattern). Lanes 0-15 hold
// column N=lane, K={0..7,16..23}; lanes 16-31 hold K={8..15,24..31}.
__device__ __forceinline__ v16h load_frag_b(const _Float16* base, int ld) {
  int lane = threadIdx.x & 31;
  int grp  = lane >> 4;
  int n    = lane & 15;
  const _Float16* p = base + (size_t)n * ld + grp * 8;
  v16h f;
  ((uint4*)&f)[0] = *(const uint4*)(p);
  ((uint4*)&f)[1] = *(const uint4*)(p + 16);
  return f;
}

__device__ __forceinline__ v8f wmma_acc(v16h a, v16h b, v8f c) {
  return __builtin_amdgcn_wmma_f32_16x16x32_f16(
      /*neg_a=*/false, a, /*neg_b=*/false, b,
      /*c_mod=*/(short)0, c, /*reuse_a=*/false, /*reuse_b=*/false);
}

// Async global -> LDS 16B copy (VGLOBAL op 98, tracked by ASYNCcnt).
// lds_off: byte address within the workgroup's LDS allocation (flat LDS
// addresses truncate to [31:0] per ISA 10.2); gaddr: 64-bit global address.
__device__ __forceinline__ void async_copy_b128(unsigned lds_off, const void* gaddr) {
  asm volatile("global_load_async_to_lds_b128 %0, %1, off"
               :: "v"(lds_off), "v"(gaddr) : "memory");
}
__device__ __forceinline__ void wait_async0() {
  asm volatile("s_wait_asynccnt 0" ::: "memory");
}

// ------------------------------ convert f32->f16 ----------------------------
__global__ void k_cvt_f16(const float* __restrict__ in,
                          _Float16* __restrict__ out, size_t n) {
  size_t i = (size_t)blockIdx.x * blockDim.x + threadIdx.x;
  if (i < n) out[i] = (_Float16)in[i];
}

// ------------------------------ generic GEMM (x @ W^T) ----------------------
// C[M,N] (f32) = A[M,K] (f16, row-major) * W[N,K]^T (f16, row-major N x K)
// Workgroup = 4 waves = 128(M) x 64(N) tile. The 64x32 weight tile is staged
// into LDS with double-buffered async copies shared by all waves; each wave
// accumulates a 32x64 tile (2 A-frags x 4 LDS B-frags -> 8 WMMA per k-step).
__global__ __launch_bounds__(128) void k_gemm_nt(
    const _Float16* __restrict__ A, const _Float16* __restrict__ W,
    float* __restrict__ C, int M, int N, int K) {
  __shared__ __align__(16) _Float16 Wlds[2][64 * 32];  // [buf][n][k], ld=32
  int tid  = threadIdx.x;
  int wave = tid >> 5;
  int n0 = blockIdx.x * 64;
  int m0 = blockIdx.y * 128 + wave * 32;

  unsigned ldsb[2];
  ldsb[0] = (unsigned)(uintptr_t)(&Wlds[0][0]);
  ldsb[1] = (unsigned)(uintptr_t)(&Wlds[1][0]);

  // Stage a 64(N) x 32(K) weight tile: 256 16B chunks over 128 threads.
  auto stage = [&](unsigned lbase, int k0) {
    for (int c = tid; c < 256; c += 128) {
      int n  = c >> 2;                  // 0..63
      int cc = c & 3;                   // 16B chunk within the 64B row
      const _Float16* g = W + (size_t)(n0 + n) * K + k0 + cc * 8;
      async_copy_b128(lbase + (unsigned)(n * 32 + cc * 8) * 2u, g);
    }
  };

  stage(ldsb[0], 0);
  wait_async0();
  __syncthreads();

  v8f acc[2][4] = {};
  const _Float16* a0 = A + (size_t)m0 * K;
  const _Float16* a1 = A + (size_t)(m0 + 16) * K;

  int nbuf = 0;
  for (int k0 = 0; k0 < K; k0 += 32) {
    bool more = (k0 + 32) < K;
    if (more) stage(ldsb[nbuf ^ 1], k0 + 32);   // prefetch next tile

    const _Float16* wl = &Wlds[nbuf][0];
    v16h af0 = load_frag_a(a0 + k0, K);
    v16h af1 = load_frag_a(a1 + k0, K);
    for (int t = 0; t < 4; ++t) {
      v16h b = load_frag_b(wl + t * 16 * 32, 32);
      acc[0][t] = wmma_acc(af0, b, acc[0][t]);
      acc[1][t] = wmma_acc(af1, b, acc[1][t]);
    }

    if (more) wait_async0();  // our async writes into next buffer are done
    __syncthreads();          // everyone finished reading cur / writing next
    nbuf ^= 1;
  }

  // C layout: lanes 0-15 hold N=lane, M=vgpr; lanes 16-31 hold M=8+vgpr.
  int lane = tid & 31;
  int grp  = lane >> 4;
  int col  = lane & 15;
  for (int half = 0; half < 2; ++half) {
    for (int i = 0; i < 8; ++i) {
      int row = m0 + half * 16 + grp * 8 + i;
      float* cp = C + (size_t)row * N + n0 + col;
      cp[0]  = acc[half][0][i];
      cp[16] = acc[half][1][i];
      cp[32] = acc[half][2][i];
      cp[48] = acc[half][3][i];
    }
  }
}

// ------------------------------ RMSNorm (f32 -> f16) ------------------------
__global__ void k_rmsnorm(const float* __restrict__ x,
                          const float* __restrict__ w,
                          _Float16* __restrict__ out, int cols, int ldin) {
  int row = blockIdx.x;
  const float* xr = x + (size_t)row * ldin;
  __shared__ float red[256];
  float s = 0.f;
  for (int i = threadIdx.x; i < cols; i += 256) {
    float v = xr[i];
    s += v * v;
  }
  red[threadIdx.x] = s;
  __syncthreads();
  for (int off = 128; off > 0; off >>= 1) {
    if ((int)threadIdx.x < off) red[threadIdx.x] += red[threadIdx.x + off];
    __syncthreads();
  }
  float r = rsqrtf(red[0] / (float)cols + 1e-6f);
  for (int i = threadIdx.x; i < cols; i += 256)
    out[(size_t)row * cols + i] = (_Float16)(w[i] * (xr[i] * r));
}

// --------------------- RoPE + Q/K/V assembly (per s, per head) --------------
// q32:  [S][H*192]  (128 nope + 64 interleaved rope, raw)
// kv32: [S][H*256]  (128 k_nope + 128 v)
// ckv32:[S][576]    (cols 512..575 = raw interleaved k_pe)
// Outputs: Qb/Kb [H][S][192] f16 (rope deinterleaved+rotated),
//          Vt    [H][128][S] f16 (V transposed, s contiguous)
__global__ void k_prep_qkv(const float* __restrict__ q32,
                           const float* __restrict__ kv32,
                           const float* __restrict__ ckv32,
                           _Float16* __restrict__ Qb,
                           _Float16* __restrict__ Kb,
                           _Float16* __restrict__ Vt) {
  int s = blockIdx.x;
  int h = blockIdx.y;
  int t = threadIdx.x;   // 64 threads
  const float* qrow  = q32  + (size_t)s * (NHEAD * QHD) + h * QHD;
  const float* kvrow = kv32 + (size_t)s * (NHEAD * 256) + h * 256;
  _Float16* Qr = Qb + ((size_t)h * S_LEN + s) * QHD;
  _Float16* Kr = Kb + ((size_t)h * S_LEN + s) * QHD;
  _Float16* Vh = Vt + (size_t)h * V_D * S_LEN;
  for (int i = t; i < NOPE_D; i += 64) {
    Qr[i] = (_Float16)qrow[i];
    Kr[i] = (_Float16)kvrow[i];
    Vh[(size_t)i * S_LEN + s] = (_Float16)kvrow[NOPE_D + i];
  }
  if (t < 32) {
    float invf = __powf(50000.f, -(float)t / 32.f);
    float ang  = (float)s * invf;
    float c = __cosf(ang), sn = __sinf(ang);
    float a = qrow[NOPE_D + 2 * t];
    float b = qrow[NOPE_D + 2 * t + 1];
    Qr[NOPE_D + t]      = (_Float16)(a * c - b * sn);
    Qr[NOPE_D + 32 + t] = (_Float16)(b * c + a * sn);
    const float* kp = ckv32 + (size_t)s * (KV_LORA + ROPE_D) + KV_LORA;
    float ka = kp[2 * t], kb = kp[2 * t + 1];
    Kr[NOPE_D + t]      = (_Float16)(ka * c - kb * sn);
    Kr[NOPE_D + 32 + t] = (_Float16)(kb * c + ka * sn);
  }
}

// ------------------------- flash attention (causal) -------------------------
// 1 wave per (16 query rows, head). Streams 32 keys/iteration:
//  scores (2x6 WMMA) -> mask -> online softmax -> P via LDS -> PV (8 WMMA).
__global__ __launch_bounds__(32) void k_mla_attn(
    const _Float16* __restrict__ Qb, const _Float16* __restrict__ Kb,
    const _Float16* __restrict__ Vt, _Float16* __restrict__ Ob) {
  int qt = blockIdx.x;
  int h  = blockIdx.y;
  int lane = threadIdx.x & 31;
  int grp  = lane >> 4;
  int ln   = lane & 15;
  const _Float16* Qh = Qb + (size_t)h * S_LEN * QHD;
  const _Float16* Kh = Kb + (size_t)h * S_LEN * QHD;
  const _Float16* Vh = Vt + (size_t)h * V_D * S_LEN;

  __shared__ __align__(16) _Float16 P[16 * 32];

  v16h qf[6];
  for (int kk = 0; kk < 6; ++kk)
    qf[kk] = load_frag_a(Qh + (size_t)qt * 16 * QHD + kk * 32, QHD);

  v8f acc[8] = {};
  float mrow[8], lrow[8];
  for (int i = 0; i < 8; ++i) { mrow[i] = -1e30f; lrow[i] = 0.f; }
  const float rscale = 0.07216878364870322f;  // 192^-0.5

  int jmax = qt * 16 + 15;
  for (int j0 = 0; j0 <= jmax; j0 += 32) {
    v8f s0 = {}, s1 = {};
    for (int kk = 0; kk < 6; ++kk) {
      v16h b0 = load_frag_b(Kh + (size_t)j0 * QHD + kk * 32, QHD);
      v16h b1 = load_frag_b(Kh + (size_t)(j0 + 16) * QHD + kk * 32, QHD);
      s0 = wmma_acc(qf[kk], b0, s0);
      s1 = wmma_acc(qf[kk], b1, s1);
    }
    float scl[8];
    for (int i = 0; i < 8; ++i) {
      int row = qt * 16 + grp * 8 + i;
      float a = s0[i] * rscale;
      float b = s1[i] * rscale;
      if (j0 + ln > row)      a = -1e30f;   // causal mask
      if (j0 + 16 + ln > row) b = -1e30f;
      float v = fmaxf(a, b);
      for (int off = 1; off < 16; off <<= 1)
        v = fmaxf(v, __shfl_xor(v, off, 32));
      float mnew = fmaxf(mrow[i], v);
      float sc = __expf(mrow[i] - mnew);
      mrow[i] = mnew;
      scl[i] = sc;
      float p0 = __expf(a - mnew);
      float p1 = __expf(b - mnew);
      float ps = p0 + p1;
      for (int off = 1; off < 16; off <<= 1)
        ps += __shfl_xor(ps, off, 32);
      lrow[i] = lrow[i] * sc + ps;
      P[(grp * 8 + i) * 32 + ln]      = (_Float16)p0;
      P[(grp * 8 + i) * 32 + 16 + ln] = (_Float16)p1;
    }
    for (int t = 0; t < 8; ++t)
      for (int i = 0; i < 8; ++i) acc[t][i] *= scl[i];
    // C-layout stores above, A-layout loads below: drain LDS (CDNA5 counter)
    asm volatile("s_wait_dscnt 0" ::: "memory");
    v16h pa = load_frag_a(P, 32);
    for (int t = 0; t < 8; ++t) {
      v16h vb = load_frag_b(Vh + (size_t)(t * 16) * S_LEN + j0, S_LEN);
      acc[t] = wmma_acc(pa, vb, acc[t]);
    }
  }
  for (int i = 0; i < 8; ++i) {
    int row = qt * 16 + grp * 8 + i;
    float inv = 1.0f / lrow[i];
    _Float16* orow = Ob + (size_t)row * (NHEAD * V_D) + h * V_D;
    for (int t = 0; t < 8; ++t)
      orow[t * 16 + ln] = (_Float16)(acc[t][i] * inv);
  }
}

// ---------------------------------------------------------------------------
extern "C" void kernel_launch(void* const* d_in, const int* in_sizes, int n_in,
                              void* d_out, int out_size, void* d_ws, size_t ws_size,
                              hipStream_t stream) {
  (void)in_sizes; (void)n_in; (void)out_size; (void)ws_size;
  const float* hidden   = (const float*)d_in[0];
  // d_in[1] = position_ids (arange; positions computed analytically)
  const float* q_a_w    = (const float*)d_in[2];
  const float* q_a_ln   = (const float*)d_in[3];
  const float* q_b_w    = (const float*)d_in[4];
  const float* kv_a_w   = (const float*)d_in[5];
  const float* kv_a_ln  = (const float*)d_in[6];
  const float* kv_b_w   = (const float*)d_in[7];
  const float* o_w      = (const float*)d_in[8];
  float* out = (float*)d_out;

  char* ws = (char*)d_ws;
  size_t off = 0;
  auto alloc = [&](size_t bytes) -> void* {
    void* p = ws + off;
    off = (off + bytes + 255) & ~(size_t)255;
    return p;
  };

  const int S = S_LEN, D = D_MODEL, QL = Q_LORA, KVL = KV_LORA;
  const int H = NHEAD, RO = ROPE_D;

  _Float16* h16    = (_Float16*)alloc((size_t)S * D * 2);
  _Float16* qaw16  = (_Float16*)alloc((size_t)QL * D * 2);
  _Float16* qbw16  = (_Float16*)alloc((size_t)H * QHD * QL * 2);
  _Float16* kvaw16 = (_Float16*)alloc((size_t)(KVL + RO) * D * 2);
  _Float16* kvbw16 = (_Float16*)alloc((size_t)H * 256 * KVL * 2);
  _Float16* ow16   = (_Float16*)alloc((size_t)D * (H * V_D) * 2);
  float*    qlat32 = (float*)alloc((size_t)S * QL * 4);
  _Float16* qlat16 = (_Float16*)alloc((size_t)S * QL * 2);
  float*    ckv32  = (float*)alloc((size_t)S * (KVL + RO) * 4);
  _Float16* ckv16  = (_Float16*)alloc((size_t)S * KVL * 2);
  float*    q32    = (float*)alloc((size_t)S * H * QHD * 4);
  float*    kv32   = (float*)alloc((size_t)S * H * 256 * 4);
  _Float16* Qb     = (_Float16*)alloc((size_t)H * S * QHD * 2);
  _Float16* Kb     = (_Float16*)alloc((size_t)H * S * QHD * 2);
  _Float16* Vt     = (_Float16*)alloc((size_t)H * V_D * S * 2);
  _Float16* attn16 = (_Float16*)alloc((size_t)S * (H * V_D) * 2);

  auto cvt = [&](const float* src, _Float16* dst, size_t n) {
    k_cvt_f16<<<(unsigned)((n + 255) / 256), 256, 0, stream>>>(src, dst, n);
  };
  cvt(hidden, h16,    (size_t)S * D);
  cvt(q_a_w,  qaw16,  (size_t)QL * D);
  cvt(q_b_w,  qbw16,  (size_t)H * QHD * QL);
  cvt(kv_a_w, kvaw16, (size_t)(KVL + RO) * D);
  cvt(kv_b_w, kvbw16, (size_t)H * 256 * KVL);
  cvt(o_w,    ow16,   (size_t)D * H * V_D);

  // q latent: [S,QL] = hidden @ q_a_w^T
  k_gemm_nt<<<dim3(QL / 64, S / 128), 128, 0, stream>>>(h16, qaw16, qlat32, S, QL, D);
  // ckv: [S,KVL+ROPE] = hidden @ kv_a_w^T
  k_gemm_nt<<<dim3((KVL + RO) / 64, S / 128), 128, 0, stream>>>(h16, kvaw16, ckv32, S, KVL + RO, D);

  // RMSNorm latents -> f16
  k_rmsnorm<<<S, 256, 0, stream>>>(qlat32, q_a_ln, qlat16, QL, QL);
  k_rmsnorm<<<S, 256, 0, stream>>>(ckv32, kv_a_ln, ckv16, KVL, KVL + RO);

  // q: [S, H*192] = qlat16 @ q_b_w^T
  k_gemm_nt<<<dim3((H * QHD) / 64, S / 128), 128, 0, stream>>>(qlat16, qbw16, q32, S, H * QHD, QL);
  // kv: [S, H*256] = ckv16 @ kv_b_w^T
  k_gemm_nt<<<dim3((H * 256) / 64, S / 128), 128, 0, stream>>>(ckv16, kvbw16, kv32, S, H * 256, KVL);

  // RoPE + per-head Q/K/V assembly (V transposed)
  k_prep_qkv<<<dim3(S, H), 64, 0, stream>>>(q32, kv32, ckv32, Qb, Kb, Vt);

  // causal flash attention
  k_mla_attn<<<dim3(S / 16, H), 32, 0, stream>>>(Qb, Kb, Vt, attn16);

  // output projection: [S,D] = attn @ o_w^T
  k_gemm_nt<<<dim3(D / 64, S / 16), 32, 0, stream>>>(attn16, ow16, out, S, D, H * V_D);
}
